// VectorQuantizer_34059090657296
// MI455X (gfx1250) — matrix operand order repaired and checked by reading, not compiled, not run on hardware
//
#include <hip/hip_runtime.h>
#include <cstdint>
#include <cstddef>

#define KC 1024
#define DC 512
#define NC 65536
#define DECAYF 0.99f
#define EPSV 1e-5f

typedef _Float16 v16h __attribute__((ext_vector_type(16)));
typedef _Float16 v8h  __attribute__((ext_vector_type(8)));
typedef float    v8f  __attribute__((ext_vector_type(8)));
typedef unsigned int v4u __attribute__((ext_vector_type(4)));
typedef int v8i __attribute__((ext_vector_type(8)));
typedef int v4i __attribute__((ext_vector_type(4)));

// ---- geometry -------------------------------------------------------------
#define XP      520      // padded row pitch in halves (512 + 8) -> 1040 B (16B aligned)
#define TILE_H  8320     // halves per codebook tile: 16 rows * 520
#define TILE_B  16640    // bytes per codebook tile
#define SLOT_B  33280    // hi tile + lo tile

// ---- LDS map (bytes) ------------------------------------------------------
#define L_XH   0
#define L_XL   66560
#define L_CB   133120    // 4 slots * 33280 = 133120
#define L_MIN  266240
#define L_IDX  266752
#define L_FIN  267264
#define LDS_TOTAL 267520

// ---- workspace map (bytes) ------------------------------------------------
#define W_CNT 0
#define W_BSQ 4096
#define W_SUM 8192
#define W_CBH 2105344
#define W_CBL 3170304

// ---- output map (float elements) ------------------------------------------
#define O_ZQ  0ull
#define O_CS  33554432ull
#define O_AVG 33555456ull
#define O_NCB 34079744ull
#define O_IDX 34604032ull

// ---------------------------------------------------------------------------
union HU { v16h v; v8h h[2]; };

__device__ __forceinline__ v16h ld_frag(const _Float16* p0, const _Float16* p1) {
  HU u;
  u.h[0] = *(const v8h*)p0;
  u.h[1] = *(const v8h*)p1;
  return u.v;
}

__device__ __forceinline__ v8f wmma16(v16h a, v16h b, v8f c) {
  return __builtin_amdgcn_wmma_f32_16x16x32_f16(false, a, false, b, (short)0, c, false, false);
}

// TDM: 1D contiguous copy of TILE_H 2-byte elements, global -> LDS.
__device__ __forceinline__ void tdm_issue(unsigned lds_byte, unsigned long long gaddr) {
  v4u g0;
  g0[0] = 1u;                                               // count=1, user desc, no gather
  g0[1] = lds_byte;                                         // lds_addr
  g0[2] = (unsigned)(gaddr & 0xffffffffull);                // global_addr[31:0]
  g0[3] = (unsigned)((gaddr >> 32) & 0x01ffffffull) | (2u << 30); // ga[56:32], type=2
  v8i g1;
  g1[0] = 0x00010000;                                       // data_size = 2 bytes
  g1[1] = (int)((TILE_H & 0xffff) << 16);                   // tensor_dim0[15:0]
  g1[2] = (int)(((TILE_H >> 16) & 0xffff) | (1 << 16));     // tensor_dim0[31:16] | tensor_dim1=1
  g1[3] = (int)((TILE_H & 0xffff) << 16);                   // tile_dim0 = TILE_H
  g1[4] = 1;                                                // tile_dim1=1, tile_dim2=0
  g1[5] = TILE_H;                                           // tensor_dim0_stride[31:0]
  g1[6] = (int)((TILE_H & 0xffff) << 16);                   // dim1_stride low16 (unused)
  g1[7] = 0;
  v4i z4 = {0, 0, 0, 0};
#if defined(__clang_major__) && (__clang_major__ >= 23)
  v8i z8 = {0, 0, 0, 0, 0, 0, 0, 0};
  __builtin_amdgcn_tensor_load_to_lds(g0, g1, z4, z4, z8, 0);
#else
  __builtin_amdgcn_tensor_load_to_lds(g0, g1, z4, z4, 0);
#endif
}

// ---------------------------------------------------------------------------
__global__ void vq_zero(float* __restrict__ p, int n) {
  int g = blockIdx.x * 256 + threadIdx.x;
  if (g < n) p[g] = 0.f;
}

__global__ void vq_bsq(const float* __restrict__ cb, float* __restrict__ bsq) {
  int k = blockIdx.x * 256 + threadIdx.x;
  float a = 0.f;
  for (int d = 0; d < DC; ++d) { float v = cb[(size_t)d * KC + k]; a = fmaf(v, v, a); }
  bsq[k] = a;
}

// codebook (D,K) f32 -> hi/lo f16 transposed to [k*XP + d]
__global__ void vq_convert(const float* __restrict__ cb,
                           _Float16* __restrict__ cbh, _Float16* __restrict__ cbl) {
  int g = blockIdx.x * 256 + threadIdx.x;   // g = d*K + k, coalesced read
  int k = g & (KC - 1);
  int d = g >> 10;
  float v = cb[g];
  _Float16 h = (_Float16)v;
  cbh[(size_t)k * XP + d] = h;
  cbl[(size_t)k * XP + d] = (_Float16)(v - (float)h);
}

// ---------------------------------------------------------------------------
__global__ __launch_bounds__(256, 1)
void vq_main(const float* __restrict__ x, const float* __restrict__ codebook,
             const float* __restrict__ bsq,
             const _Float16* __restrict__ cbh_g, const _Float16* __restrict__ cbl_g,
             float* __restrict__ counts, float* __restrict__ sum_dk,
             float* __restrict__ zq, float* __restrict__ oidx)
{
  extern __shared__ char smem[];
  _Float16* xh = (_Float16*)(smem + L_XH);
  _Float16* xl = (_Float16*)(smem + L_XL);
  float*  lmin = (float*)(smem + L_MIN);
  int*    lidx = (int*)(smem + L_IDX);
  int*    ifin = (int*)(smem + L_FIN);

  const int tid  = threadIdx.x;
  const int lane = tid & 31;
  const int w    = tid >> 5;      // wave 0..7
  const int ng   = w & 3;         // n-group (16 rows each)
  const int kh   = w >> 2;        // k-half (512 codes each)
  const int col  = lane & 15;
  const int hi   = lane >> 4;
  const int n0   = blockIdx.x * 64;
  const bool issuer = ((w & 3) == 0);   // waves 0 and 4 drive TDM for their k-half

  const unsigned cb_lds0 = (unsigned)(size_t)(smem + L_CB);
  const unsigned long long gh = (unsigned long long)(size_t)cbh_g;
  const unsigned long long gl = (unsigned long long)(size_t)cbl_g;

  // Prologue: prefetch first two codebook tiles per k-half via TDM.
  if (issuer) {
#pragma unroll
    for (int p = 0; p < 2; ++p) {
      const int kt = kh * 32 + p;
      const unsigned slot = cb_lds0 + (unsigned)(kh * 2 + p) * SLOT_B;
      tdm_issue(slot,          gh + (unsigned long long)kt * TILE_B);
      tdm_issue(slot + TILE_B, gl + (unsigned long long)kt * TILE_B);
    }
  }

  // Stage x tile (64 columns) as split hi/lo f16, [n][d] with padded pitch.
  {
    const int nl = tid & 63;
    const int n  = n0 + nl;
    for (int d = tid >> 6; d < DC; d += 4) {
      float v = x[(size_t)d * NC + n];
      _Float16 h = (_Float16)v;
      xh[nl * XP + d] = h;
      xl[nl * XP + d] = (_Float16)(v - (float)h);
    }
  }

  float rmin[8]; int ridx[8];
#pragma unroll
  for (int r = 0; r < 8; ++r) { rmin[r] = 3.4e38f; ridx[r] = 0; }

  // A-fragment base: lane holds row M=col of its n-group; K split per A layout.
  const _Float16* axh = xh + (ng * 16 + col) * XP + (hi ? 8 : 0);
  const _Float16* axl = xl + (ng * 16 + col) * XP + (hi ? 8 : 0);

  for (int s = 0; s < 32; ++s) {
    const int kt = kh * 32 + s;
    if (issuer) {
      if (s == 31) __builtin_amdgcn_s_wait_tensorcnt((short)0);
      else         __builtin_amdgcn_s_wait_tensorcnt((short)2);
    }
    __syncthreads();

    const _Float16* bh =
        (const _Float16*)(smem + L_CB + (kh * 2 + (s & 1)) * SLOT_B) + col * XP + hi * 8;
    const _Float16* bl = bh + TILE_H;

    v8f C = {0.f, 0.f, 0.f, 0.f, 0.f, 0.f, 0.f, 0.f};
#pragma unroll
    for (int c = 0; c < 16; ++c) {
      const int db = c * 32;
      v16h Ah = ld_frag(axh + db, axh + db + 16);
      v16h Al = ld_frag(axl + db, axl + db + 16);
      v16h Bh = ld_frag(bh + db, bh + db + 16);
      v16h Bl = ld_frag(bl + db, bl + db + 16);
      C = wmma16(Ah, Bh, C);   // hi*hi
      C = wmma16(Ah, Bl, C);   // hi*lo
      C = wmma16(Al, Bh, C);   // lo*hi
    }

    const int kg = kt * 16 + col;
    const float bq = bsq[kg];
#pragma unroll
    for (int r = 0; r < 8; ++r) {
      float sc = fmaf(-2.f, C[r], bq);       // ||c||^2 - 2 x.c  (||x||^2 const per n)
      if (sc < rmin[r]) { rmin[r] = sc; ridx[r] = kg; }
    }

    __syncthreads();
    if (issuer && s < 30) {                  // prefetch tile s+2 into the freed slot
      const int kt2 = kh * 32 + s + 2;
      const unsigned slot = cb_lds0 + (unsigned)(kh * 2 + (s & 1)) * SLOT_B;
      tdm_issue(slot,          gh + (unsigned long long)kt2 * TILE_B);
      tdm_issue(slot + TILE_B, gl + (unsigned long long)kt2 * TILE_B);
    }
  }

  // Butterfly min-reduce across the 16 k-columns held by each half-wave.
#pragma unroll
  for (int m = 1; m <= 8; m <<= 1) {
#pragma unroll
    for (int r = 0; r < 8; ++r) {
      float om = __shfl_xor(rmin[r], m, 32);
      int   oi = __shfl_xor(ridx[r], m, 32);
      if (om < rmin[r] || (om == rmin[r] && oi < ridx[r])) { rmin[r] = om; ridx[r] = oi; }
    }
  }
  if (col == 0) {
    const int base = ng * 16 + hi * 8;
#pragma unroll
    for (int r = 0; r < 8; ++r) {
      lmin[kh * 64 + base + r] = rmin[r];
      lidx[kh * 64 + base + r] = ridx[r];
    }
  }
  __syncthreads();

  if (tid < 64) {           // merge the two k-halves, emit index + count
    float m0 = lmin[tid], m1 = lmin[64 + tid];
    int   i0 = lidx[tid], i1 = lidx[64 + tid];
    int best = (m1 < m0 || (m1 == m0 && i1 < i0)) ? i1 : i0;
    ifin[tid] = best;
    atomicAdd(counts + best, 1.0f);
    oidx[n0 + tid] = (float)best;
  }
  __syncthreads();

  // Fused epilogue: z_q gather + segment-sum scatter (x reconstructed hi+lo).
  {
    const int nl = tid & 63;
    const int n  = n0 + nl;
    const int ci = ifin[nl];
    for (int d = tid >> 6; d < DC; d += 4) {
      float xv = (float)xh[nl * XP + d] + (float)xl[nl * XP + d];
      zq[(size_t)d * NC + n] = codebook[(size_t)d * KC + ci];
      atomicAdd(sum_dk + (size_t)d * KC + ci, xv);
    }
  }
}

// ---------------------------------------------------------------------------
__global__ void vq_f1(const float* __restrict__ cs, const float* __restrict__ counts,
                      float* __restrict__ ocs) {
  __shared__ float red[1024];
  int k = threadIdx.x;
  float v = DECAYF * cs[k] + (1.f - DECAYF) * counts[k];
  red[k] = v;
  __syncthreads();
  for (int off = 512; off > 0; off >>= 1) {
    if (k < off) red[k] += red[k + off];
    __syncthreads();
  }
  float n = red[0];
  ocs[k] = (v + EPSV) / (n + KC * EPSV) * n;
}

__global__ void vq_f2(const float* __restrict__ cba, const float* __restrict__ sumdk,
                      const float* __restrict__ ncs,
                      float* __restrict__ oavg, float* __restrict__ oncb) {
  int g = blockIdx.x * 256 + threadIdx.x;   // over D*K, layout [d*K + k]
  int k = g & (KC - 1);
  float a = cba[g];
  oavg[g] = DECAYF * a + (1.f - DECAYF) * sumdk[g];
  oncb[g] = a / ncs[k];                     // faithful: OLD codebook_avg / new_cluster_size
}

// ---------------------------------------------------------------------------
extern "C" void kernel_launch(void* const* d_in, const int* in_sizes, int n_in,
                              void* d_out, int out_size, void* d_ws, size_t ws_size,
                              hipStream_t stream) {
  const float* x   = (const float*)d_in[0];
  const float* cb  = (const float*)d_in[1];
  const float* cba = (const float*)d_in[2];
  const float* cs  = (const float*)d_in[3];
  float* out = (float*)d_out;
  char*  ws  = (char*)d_ws;

  float*    counts = (float*)(ws + W_CNT);
  float*    bsq    = (float*)(ws + W_BSQ);
  float*    sumdk  = (float*)(ws + W_SUM);
  _Float16* cbh    = (_Float16*)(ws + W_CBH);
  _Float16* cbl    = (_Float16*)(ws + W_CBL);

  (void)in_sizes; (void)n_in; (void)out_size; (void)ws_size;
  hipFuncSetAttribute((const void*)vq_main,
                      hipFuncAttributeMaxDynamicSharedMemorySize, LDS_TOTAL);

  vq_zero<<<2056, 256, 0, stream>>>((float*)ws, 526336);     // counts + bsq + sum_dk
  vq_bsq<<<4, 256, 0, stream>>>(cb, bsq);
  vq_convert<<<2048, 256, 0, stream>>>(cb, cbh, cbl);
  vq_main<<<1024, 256, LDS_TOTAL, stream>>>(x, cb, bsq, cbh, cbl,
                                            counts, sumdk,
                                            out + O_ZQ, out + O_IDX);
  vq_f1<<<1, 1024, 0, stream>>>(cs, counts, out + O_CS);
  vq_f2<<<2048, 256, 0, stream>>>(cba, sumdk, out + O_CS, out + O_AVG, out + O_NCB);
}
